// RecurrentNetworkModel_85426899517832
// MI455X (gfx1250) — compile-verified, matrix operand order checked by reading, not run on hardware
//
#include <hip/hip_runtime.h>
#include <hip/hip_bf16.h>
#include <stdint.h>
#include <stddef.h>

// ---------------------------------------------------------------------------
// Problem constants (from reference): S=512, B=64, IN=512, H=1024, OUT=512
// ---------------------------------------------------------------------------
#define S_LEN   512
#define BATCH   64
#define IN_DIM  512
#define H_DIM   1024
#define OUT_DIM 512
#define G_DIM   4096   // 4*H

typedef __bf16 bf16;
typedef __attribute__((ext_vector_type(16))) __bf16 v16bf;
typedef __attribute__((ext_vector_type(8)))  float  v8f;

#define WMMA_BF16(a, b, c) \
    __builtin_amdgcn_wmma_f32_16x16x32_bf16(false, (a), false, (b), (short)0, (c), false, false)

// float -> bf16 (round to nearest even)
__device__ __forceinline__ bf16 to_bf16(float f) {
    union { float f; unsigned u; } v; v.f = f;
    unsigned r = v.u + 0x7fffu + ((v.u >> 16) & 1u);
    unsigned short h = (unsigned short)(r >> 16);
    bf16 b;
    __builtin_memcpy(&b, &h, 2);
    return b;
}

__device__ __forceinline__ float sigm(float x) {
    return 1.0f / (1.0f + __expf(-x));
}
__device__ __forceinline__ float tanh_fast(float x) {
    x = fminf(fmaxf(x, -15.0f), 15.0f);
    float e = __expf(2.0f * x);
    return (e - 1.0f) / (e + 1.0f);
}

// ---------------------------------------------------------------------------
// Swizzle mapping (WMMA bf16 16x16x32 operand register layout, wave32):
//   element e (0..15) of lane L holds K' = (e&7) + ((e>>3)<<4) + (L>=16 ? 8:0)
//   A fragment: row M = L%16 ; B fragment: col N = L%16.
// Inverse: given K' in [0,32):  hi=(K'>>3)&1 ; e=(K'&7)|((K'>>4)<<3)
//
// A operand storage: [kt][mt(4)][512 elems]  -> mt stride = 512 elems (1KB),
//   kt stride = 2048 elems (4KB); the 4 M-fragments of one k-tile are one
//   contiguous 4KB region. Identical formula for x, h0, h1 blocks.
// B operand storage: [nt][kt][512 elems].
// ---------------------------------------------------------------------------

// Branch-free K-range MMA: 4 M-tiles x 1 N-tile, A at [kt][mt] layout.
__device__ __forceinline__ void mma_krange(const bf16* __restrict__ Ab,
                                           const bf16* __restrict__ Bb,
                                           int nkt,
                                           v8f& acc0, v8f& acc1, v8f& acc2, v8f& acc3)
{
#pragma unroll 2
    for (int kt = 0; kt < nkt; ++kt) {
        v16bf bf = *(const v16bf*)(Bb + (size_t)kt * 512);
        const bf16* Ap = Ab + (size_t)kt * 2048;
        v16bf a0 = *(const v16bf*)(Ap);
        v16bf a1 = *(const v16bf*)(Ap + 512);
        v16bf a2 = *(const v16bf*)(Ap + 1024);
        v16bf a3 = *(const v16bf*)(Ap + 1536);
        __builtin_prefetch(Bb + (size_t)(kt + 1) * 512, 0, 1);  // speculative: safe past end
        acc0 = WMMA_BF16(a0, bf, acc0);
        acc1 = WMMA_BF16(a1, bf, acc1);
        acc2 = WMMA_BF16(a2, bf, acc2);
        acc3 = WMMA_BF16(a3, bf, acc3);
    }
}

// Pack x[S][B][IN] fp32 -> x_swz[s][kt(16)][mt(4)][512] bf16 (A-fragment layout)
__global__ void pack_x_kernel(const float* __restrict__ x, bf16* __restrict__ xs, int total) {
    int tid = blockIdx.x * 256 + threadIdx.x;
    if (tid >= total) return;
    int s   = tid / (BATCH * IN_DIM);
    int rem = tid % (BATCH * IN_DIM);
    int b = rem / IN_DIM;
    int k = rem % IN_DIM;
    int mt = b >> 4, kt = k >> 5, kp = k & 31;
    int hi = (kp >> 3) & 1;
    int e  = (kp & 7) | ((kp >> 4) << 3);
    int ln = (b & 15) + hi * 16;
    xs[(size_t)s * 32768 + (size_t)(kt * 4 + mt) * 512 + ln * 16 + e] = to_bf16(x[tid]);
}

// Pack concat(Wx^T, Uh^T) [K x G] -> B_swz[nt(G/16)][kt(Kt)][512] (B-fragment layout)
// Bm[k][g] = k < KX ? Wx[g][k] : Uh[g][k-KX]
__global__ void pack_w_kernel(const float* __restrict__ Wx, const float* __restrict__ Uh,
                              bf16* __restrict__ dst, int KX, int Kt, int total) {
    int tid = blockIdx.x * 256 + threadIdx.x;
    if (tid >= total) return;
    int nt = tid / (Kt * 512);
    int r  = tid % (Kt * 512);
    int kt = r / 512;
    int qq = r % 512;
    int ln = qq >> 4, e = qq & 15;
    int n  = nt * 16 + (ln & 15);
    int hi = ln >> 4;
    int kp = (e & 7) + ((e >> 3) << 4) + hi * 8;
    int k  = kt * 32 + kp;
    int KU = Kt * 32 - KX;
    float v = (k < KX) ? Wx[(size_t)n * KX + k] : Uh[(size_t)n * KU + (k - KX)];
    dst[tid] = to_bf16(v);
}

// ---------------------------------------------------------------------------
// Fused LSTM step: blocks 0..15 -> layer0 @ time s, blocks 16..31 -> layer1 @ s-1.
// Each WG owns 64 hidden columns across all 4 gate quadrants:
//   gates[64B x 256] = [A0 | A1(h_prev)] x B_swz  (K = Kt0*32 + 1024)
// 16 waves, each wave: one 64(M)x16(N) strip, 4 fp32 accumulators, 2 branch-free
// K-loops of WMMA (x/h0 part, then h_prev part). Pointwise tail applies biases +
// LSTM update, writes h in A-swizzled layout.
// h histories stored as (S+1) blocks of [32][4][512]; block 0 = zeros.
// ---------------------------------------------------------------------------
__global__ __launch_bounds__(512) void lstm_step_fused(
    int s,
    const bf16* __restrict__ x_swz,
    bf16* __restrict__ h0_all, bf16* __restrict__ h1_all,
    const bf16* __restrict__ B0, const bf16* __restrict__ B1,
    const float* __restrict__ bx0, const float* __restrict__ bh0,
    const float* __restrict__ bx1, const float* __restrict__ bh1,
    float* __restrict__ c0, float* __restrict__ c1)
{
    __shared__ float gates[64 * 256];

    const int layer = blockIdx.x >> 4;
    const int w     = blockIdx.x & 15;

    const bf16 *A0, *A1, *Bs;
    const float *bx, *bh;
    float* c;
    bf16* hout;
    int Kt0;

    if (layer == 0) {
        if (s >= S_LEN) return;
        A0  = x_swz + (size_t)s * 32768;  Kt0 = 16;
        A1  = h0_all + (size_t)s * 65536;            // h0[t-1] (block s)
        Bs  = B0; bx = bx0; bh = bh0; c = c0;
        hout = h0_all + (size_t)(s + 1) * 65536;
    } else {
        int s1 = s - 1;
        if (s1 < 0) return;
        A0  = h0_all + (size_t)s * 65536; Kt0 = 32;  // layer0 output @ s1 = block s1+1
        A1  = h1_all + (size_t)s1 * 65536;
        Bs  = B1; bx = bx1; bh = bh1; c = c1;
        hout = h1_all + (size_t)s * 65536;
    }
    const int Kt = Kt0 + 32;

    const int lane = threadIdx.x & 31;
    const int wv   = threadIdx.x >> 5;       // 0..15
    const int q    = wv >> 2;                // gate quadrant 0..3 (f,i,o,g)
    const int nsub = wv & 3;                 // 16-col subtile within WG's 64 cols
    const int nt   = q * 64 + w * 4 + nsub;  // global N-tile over 4096 gate cols

    const bf16* Bt  = Bs + ((size_t)nt * Kt) * 512 + (size_t)lane * 16;
    const bf16* A0l = A0 + (size_t)lane * 16;
    const bf16* A1l = A1 + (size_t)lane * 16;

    v8f acc0 = {0,0,0,0,0,0,0,0};
    v8f acc1 = {0,0,0,0,0,0,0,0};
    v8f acc2 = {0,0,0,0,0,0,0,0};
    v8f acc3 = {0,0,0,0,0,0,0,0};

    mma_krange(A0l, Bt,                        Kt0, acc0, acc1, acc2, acc3);
    mma_krange(A1l, Bt + (size_t)Kt0 * 512,    32,  acc0, acc1, acc2, acc3);

    // Spill accumulators to LDS: C/D layout: VGPR r, lane l -> M=r+(l<16?0:8), N=l%16
    const int colb   = q * 64 + nsub * 16 + (lane & 15);
    const int rowoff = (lane < 16) ? 0 : 8;
#pragma unroll
    for (int r = 0; r < 8; ++r) {
        gates[(0 * 16 + r + rowoff) * 256 + colb] = acc0[r];
        gates[(1 * 16 + r + rowoff) * 256 + colb] = acc1[r];
        gates[(2 * 16 + r + rowoff) * 256 + colb] = acc2[r];
        gates[(3 * 16 + r + rowoff) * 256 + colb] = acc3[r];
    }
    __syncthreads();

    // Pointwise LSTM update for this WG's 64 hidden columns (all 64 batch rows)
#pragma unroll
    for (int p = 0; p < 8; ++p) {
        int idx = (int)threadIdx.x * 8 + p;      // 0..4095
        int b  = idx >> 6;
        int jj = idx & 63;
        int j  = w * 64 + jj;
        float gf = gates[b * 256 +       jj] + bx[j]             + bh[j];
        float gi = gates[b * 256 +  64 + jj] + bx[H_DIM + j]     + bh[H_DIM + j];
        float go = gates[b * 256 + 128 + jj] + bx[2 * H_DIM + j] + bh[2 * H_DIM + j];
        float gg = gates[b * 256 + 192 + jj] + bx[3 * H_DIM + j] + bh[3 * H_DIM + j];
        float cp = c[b * H_DIM + j];
        float cn = sigm(gf) * cp + sigm(gi) * tanh_fast(gg);
        float hv = sigm(go) * tanh_fast(cn);
        c[b * H_DIM + j] = cn;
        // write h in A-fragment swizzled layout [kt][mt][512]
        int mt = b >> 4, kt = j >> 5, kp = j & 31;
        int hi = (kp >> 3) & 1;
        int e  = (kp & 7) | ((kp >> 4) << 3);
        int ln = (b & 15) + hi * 16;
        hout[(size_t)(kt * 4 + mt) * 512 + ln * 16 + e] = to_bf16(hv);
    }
}

// ---------------------------------------------------------------------------
// FC head: out[S*B=32768, 512] = h1_all (swizzled A) x Bfc + bias
// Grid: 512 WGs of 512 threads; WG g handles one timestep (64 rows = batch);
// each wave handles 2 N-tiles.
// ---------------------------------------------------------------------------
__global__ __launch_bounds__(512) void fc_kernel(
    const bf16* __restrict__ A, const bf16* __restrict__ Bw,
    const float* __restrict__ bias, float* __restrict__ out)
{
    const int lane = threadIdx.x & 31;
    const int wv   = threadIdx.x >> 5;
    const int g    = blockIdx.x;                     // timestep s
    const bf16* Al = A + (size_t)g * 65536 + (size_t)lane * 16;

    for (int st = 0; st < 2; ++st) {
        int nt = wv * 2 + st;                        // 0..31 over 512 out cols
        const bf16* Bt = Bw + ((size_t)nt * 32) * 512 + (size_t)lane * 16;
        v8f acc0 = {0,0,0,0,0,0,0,0};
        v8f acc1 = {0,0,0,0,0,0,0,0};
        v8f acc2 = {0,0,0,0,0,0,0,0};
        v8f acc3 = {0,0,0,0,0,0,0,0};

        mma_krange(Al, Bt, 32, acc0, acc1, acc2, acc3);

        const int col    = nt * 16 + (lane & 15);
        const int rowoff = (lane < 16) ? 0 : 8;
        const float bv   = bias[col];
#pragma unroll
        for (int r = 0; r < 8; ++r) {
            out[((size_t)g * 64 +  0 + r + rowoff) * OUT_DIM + col] = acc0[r] + bv;
            out[((size_t)g * 64 + 16 + r + rowoff) * OUT_DIM + col] = acc1[r] + bv;
            out[((size_t)g * 64 + 32 + r + rowoff) * OUT_DIM + col] = acc2[r] + bv;
            out[((size_t)g * 64 + 48 + r + rowoff) * OUT_DIM + col] = acc3[r] + bv;
        }
    }
}

// ---------------------------------------------------------------------------
extern "C" void kernel_launch(void* const* d_in, const int* in_sizes, int n_in,
                              void* d_out, int out_size, void* d_ws, size_t ws_size,
                              hipStream_t stream) {
    const float* x   = (const float*)d_in[0];
    const float* w0x = (const float*)d_in[1];
    const float* u0  = (const float*)d_in[2];
    const float* bx0 = (const float*)d_in[3];
    const float* bh0 = (const float*)d_in[4];
    const float* w1x = (const float*)d_in[5];
    const float* u1  = (const float*)d_in[6];
    const float* bx1 = (const float*)d_in[7];
    const float* bh1 = (const float*)d_in[8];
    const float* fcw = (const float*)d_in[9];
    const float* fcb = (const float*)d_in[10];
    float* out = (float*)d_out;

    char* p = (char*)d_ws;
    auto carve = [&](size_t bytes) -> char* {
        char* r = p;
        p += (bytes + 255) & ~(size_t)255;
        return r;
    };
    bf16*  x_swz = (bf16*)carve((size_t)S_LEN * 32768 * 2);        // 32 MB
    bf16*  B0    = (bf16*)carve((size_t)256 * 48 * 512 * 2);       // 12.6 MB
    bf16*  B1    = (bf16*)carve((size_t)256 * 64 * 512 * 2);       // 16.8 MB
    bf16*  Bfc   = (bf16*)carve((size_t)32 * 32 * 512 * 2);        //  1 MB
    bf16*  h0    = (bf16*)carve((size_t)(S_LEN + 1) * 65536 * 2);  // 64 MB
    bf16*  h1    = (bf16*)carve((size_t)(S_LEN + 1) * 65536 * 2);  // 64 MB
    float* c0    = (float*)carve((size_t)BATCH * H_DIM * 4);
    float* c1    = (float*)carve((size_t)BATCH * H_DIM * 4);

    // zero-init h0/h1 block 0 (t = -1 state) and cell states
    hipMemsetAsync(h0, 0, 65536 * 2, stream);
    hipMemsetAsync(h1, 0, 65536 * 2, stream);
    hipMemsetAsync(c0, 0, (size_t)BATCH * H_DIM * 4, stream);
    hipMemsetAsync(c1, 0, (size_t)BATCH * H_DIM * 4, stream);

    { int t = S_LEN * BATCH * IN_DIM;
      pack_x_kernel<<<(t + 255) / 256, 256, 0, stream>>>(x, x_swz, t); }
    { int t = 256 * 48 * 512;
      pack_w_kernel<<<(t + 255) / 256, 256, 0, stream>>>(w0x, u0, B0, IN_DIM, 48, t); }
    { int t = 256 * 64 * 512;
      pack_w_kernel<<<(t + 255) / 256, 256, 0, stream>>>(w1x, u1, B1, H_DIM, 64, t); }
    { int t = 32 * 32 * 512;
      pack_w_kernel<<<(t + 255) / 256, 256, 0, stream>>>(fcw, fcw, Bfc, H_DIM, 32, t); }

    // software-pipelined recurrence: layer0 @ s overlapped with layer1 @ s-1
    for (int s = 0; s <= S_LEN; ++s) {
        lstm_step_fused<<<32, 512, 0, stream>>>(s, x_swz, h0, h1, B0, B1,
                                                bx0, bh0, bx1, bh1, c0, c1);
    }

    fc_kernel<<<512, 512, 0, stream>>>(h1 + 65536, Bfc, fcb, out);
}
